// ProbAttention_50689204027551
// MI455X (gfx1250) — matrix operand order repaired and checked by reading, not compiled, not run on hardware
//
#include <hip/hip_runtime.h>
#include <math.h>

typedef __attribute__((ext_vector_type(16))) _Float16 v16h;
typedef __attribute__((ext_vector_type(8)))  float    v8f;
typedef __attribute__((ext_vector_type(4)))  float    v4f;

namespace {
constexpr int  NB = 2;      // batch
constexpr int  NL = 2048;   // sequence length
constexpr int  NH = 16;     // heads
constexpr int  ND = 64;     // head dim
constexpr int  NSL = 8;     // softmax slices per row
constexpr int  SLICE = NL / NSL; // 256
}

// One workgroup = 4 waves = 128 threads handles one (b, h, 16-query-row tile).
// LDS holds a 16 x 2048 fp32 score/probability strip (128 KB).
__global__ __launch_bounds__(128)
void ProbAttention_50689204027551_kernel(const float* __restrict__ Q,
                                         const float* __restrict__ K,
                                         const float* __restrict__ V,
                                         float* __restrict__ out,
                                         float* __restrict__ attn)
{
    extern __shared__ float smem[];
    float* S     = smem;              // [16][NL] scores -> exp -> probabilities
    float* red   = smem + 16 * NL;    // [16][NSL] partial reductions
    float* rowmx = red + 16 * NSL;    // [16]
    float* rowsm = rowmx + 16;        // [16]

    const int qt = blockIdx.x;        // query tile (16 rows)
    const int h  = blockIdx.y;
    const int b  = blockIdx.z;
    const int q0 = qt * 16;

    const int tid  = threadIdx.x;
    const int lane = tid & 31;
    const int wave = tid >> 5;
    const int half = lane >> 4;       // which 16-lane half of the wave
    const int l16  = lane & 15;

    const float scale = 0.125f;       // 1/sqrt(64)

    // ---------------- load Q tile into WMMA A-layout registers ----------------
    // A (16x32 f16): lane row M = lane%16; element i (vgpr v=i/2, pair p=i%2):
    //   K = 16*(v/4) + 8*half + 2*(v%4) + p
    v16h qa0, qa1;
    {
        const long qbase = (((long)b * NL + (q0 + l16)) * NH + h) * ND;
#pragma unroll
        for (int i = 0; i < 16; ++i) {
            const int v = i >> 1, p = i & 1;
            const int k = 16 * (v >> 2) + 8 * half + 2 * (v & 3) + p;
            qa0[i] = (_Float16)Q[qbase + k];
            qa1[i] = (_Float16)Q[qbase + 32 + k];
        }
    }

    // ---------------- S = Q * K^T over causally-active k tiles ----------------
    for (int kt = wave; kt <= qt; kt += 4) {
        const int k0 = kt * 16;
        // B (32x16 f16): lane col N = lane%16 (key index k0+N);
        //   element i (v=i/2, p=i%2): Kdim = 16*(v/4) + 8*half + 2*(v%4) + p  (head-dim index)
        v16h b0, b1;
        {
            const long kbase = (((long)b * NL + (k0 + l16)) * NH + h) * ND;
#pragma unroll
            for (int i = 0; i < 16; ++i) {
                const int v = i >> 1, p = i & 1;
                const int kd = 16 * (v >> 2) + 8 * half + 2 * (v & 3) + p;
                b0[i] = (_Float16)K[kbase + kd];
                b1[i] = (_Float16)K[kbase + 32 + kd];
            }
        }
        v8f acc = {};
        acc = __builtin_amdgcn_wmma_f32_16x16x32_f16(false, qa0, false, b0,
                                                     (short)0, acc, false, false);
        acc = __builtin_amdgcn_wmma_f32_16x16x32_f16(false, qa1, false, b1,
                                                     (short)0, acc, false, false);
        // C/D layout: vgpr j -> row M = j + 8*half, col N = lane%16
#pragma unroll
        for (int j = 0; j < 8; ++j) {
            const int m = j + 8 * half;
            S[m * NL + k0 + l16] = acc[j];
        }
    }
    __syncthreads();

    // ---------------- row softmax (16 rows x 8 slice-threads) ----------------
    const int r  = tid >> 3;                 // row 0..15
    const int s  = tid & 7;                  // slice 0..7
    const int qg = q0 + r;                   // global query index
    const int c0 = s * SLICE;
    const int limit = qg + 1;                // cols [0, limit) are valid
    const int vend  = (c0 + SLICE < limit) ? (c0 + SLICE) : limit; // may be < c0
    float* rowp = S + r * NL;

    // pass 1: slice max over valid columns (branch-free, b128 LDS reads)
    float mx = -3.0e38f;
    {
        int c = c0;
        for (; c + 4 <= vend; c += 4) {
            v4f x = *(const v4f*)(rowp + c);
            mx = fmaxf(mx, fmaxf(fmaxf(x[0], x[1]), fmaxf(x[2], x[3])));
        }
        for (; c < vend; ++c) mx = fmaxf(mx, rowp[c]);
    }
    red[r * NSL + s] = mx;
    __syncthreads();
    if (s == 0) {
        float m2 = red[r * NSL];
#pragma unroll
        for (int i = 1; i < NSL; ++i) m2 = fmaxf(m2, red[r * NSL + i]);
        rowmx[r] = m2;
    }
    __syncthreads();
    const float m = rowmx[r];

    // pass 2: e = exp(scale*(S-m)) stored back into LDS, plus slice sum
    float psum = 0.0f;
    {
        int c = c0;
        for (; c + 4 <= vend; c += 4) {
            v4f x = *(const v4f*)(rowp + c);
            v4f e;
#pragma unroll
            for (int j = 0; j < 4; ++j) e[j] = __expf(scale * (x[j] - m));
            *(v4f*)(rowp + c) = e;
            psum += (e[0] + e[1]) + (e[2] + e[3]);
        }
        for (; c < vend; ++c) {
            float e = __expf(scale * (rowp[c] - m));
            rowp[c] = e;
            psum += e;
        }
    }
    red[r * NSL + s] = psum;
    __syncthreads();
    if (s == 0) {
        float t = 0.0f;
#pragma unroll
        for (int i = 0; i < NSL; ++i) t += red[r * NSL + i];
        rowsm[r] = t;
    }
    __syncthreads();
    const float inv = 1.0f / rowsm[r];

    // pass 3: normalize, write P to LDS + global attn (non-temporal: the attn
    // stream is write-once/no-reuse; keep L2 for the K/V working set).
    float* arow = attn + (((long)b * NH + h) * NL + qg) * NL;
    {
        int c = c0;
        for (; c + 4 <= vend; c += 4) {
            v4f e = *(const v4f*)(rowp + c);
            v4f pv = e * inv;
            *(v4f*)(rowp + c) = pv;
            __builtin_nontemporal_store(pv, (v4f*)(arow + c));
        }
        for (; c < vend; ++c) {
            float pv = rowp[c] * inv;
            rowp[c] = pv;
            __builtin_nontemporal_store(pv, arow + c);
        }
        // masked region: zeros (keeps PV chunks exact, materializes attn zeros)
        for (; c < c0 + SLICE && (c & 3); ++c) {
            rowp[c] = 0.0f;
            __builtin_nontemporal_store(0.0f, arow + c);
        }
        const v4f z = {};
        for (; c + 4 <= c0 + SLICE; c += 4) {
            *(v4f*)(rowp + c) = z;
            __builtin_nontemporal_store(z, (v4f*)(arow + c));
        }
    }
    __syncthreads();

    // ---------------- out = P * V (each wave owns a 16-wide d slice) ----------
    const int d0  = wave * 16;
    const int nk  = q0 + 16;               // causally-active key count
    const int nch = (nk + 31) >> 5;        // 32-k chunks
    v8f oacc = {};
    for (int ck = 0; ck < nch; ++ck) {
        const int kc = ck * 32;
        v16h pa, vb;
#pragma unroll
        for (int i = 0; i < 16; ++i) {
            const int v = i >> 1, p = i & 1;
            const int ka = 16 * (v >> 2) + 8 * half + 2 * (v & 3) + p; // A layout
            pa[i] = (_Float16)S[l16 * NL + kc + ka];
            const int kb = 16 * (v >> 2) + 8 * half + 2 * (v & 3) + p; // B layout
            vb[i] = (_Float16)V[(((long)b * NL + (kc + kb)) * NH + h) * ND + d0 + l16];
        }
        oacc = __builtin_amdgcn_wmma_f32_16x16x32_f16(false, pa, false, vb,
                                                      (short)0, oacc, false, false);
    }
#pragma unroll
    for (int j = 0; j < 8; ++j) {
        const int mrow = j + 8 * half;
        __builtin_nontemporal_store(
            oacc[j], out + (((long)b * NL + (q0 + mrow)) * NH + h) * ND + d0 + l16);
    }
}

extern "C" void kernel_launch(void* const* d_in, const int* in_sizes, int n_in,
                              void* d_out, int out_size, void* d_ws, size_t ws_size,
                              hipStream_t stream) {
    (void)in_sizes; (void)n_in; (void)d_ws; (void)ws_size; (void)out_size;
    const float* Q = (const float*)d_in[0];
    const float* K = (const float*)d_in[1];
    const float* V = (const float*)d_in[2];
    // d_in[3] is the boolean causal mask; causality is applied analytically.

    float* out  = (float*)d_out;                       // [B, L, H, D]
    float* attn = out + (long)NB * NL * NH * ND;       // [B, H, L, L]

    dim3 grid(NL / 16, NH, NB);                        // 128 x 16 x 2 = 4096 WGs
    const size_t shmem = (size_t)(16 * NL + 16 * NSL + 16 + 16) * sizeof(float);
    ProbAttention_50689204027551_kernel<<<grid, 128, shmem, stream>>>(Q, K, V, out, attn);
}
